// MultiHeadAttention_43954695307816
// MI455X (gfx1250) — matrix operand order repaired and checked by reading, not compiled, not run on hardware
//
#include <hip/hip_runtime.h>
#include <hip/hip_bf16.h>

// ---------------------------------------------------------------------------
// Multi-head attention for MI455X (gfx1250), wave32 + v_wmma_f32_16x16x32_f16.
//   B=2, S=2048, E=1024, H=16, D=64
// Pipeline:
//   0) cvt_x        : x f32 -> Xh f16 (flat)                        [once]
//   0) transpose_w  : W f32 [k][n] -> Wt f16 [n][k] (n-major)       [once, x4]
//   1) gemm_qkv     : Xh @ Wt{q,k,v} + b -> Q,K (f16 [B,H,S,D]), Vt (f16 [B,H,D,S])
//   2) attn         : flash-style online-softmax attention, WMMA both matmuls
//   3) gemm_out     : O(f16) @ Wot + bo -> out (f32)
// GEMM kernels are LDS/barrier-free: operands are pre-packed f16 in the exact
// WMMA A/B register layouts; all packed operands are L2-resident (192 MB).
// ---------------------------------------------------------------------------

#define EMB    1024
#define HEADS  16
#define HDIM   64
#define BATCH  2
#define SEQ    2048
#define MROWS  (BATCH * SEQ)   // 4096

typedef _Float16 v16h __attribute__((ext_vector_type(16)));
typedef _Float16 v8h  __attribute__((ext_vector_type(8)));
typedef _Float16 v4h  __attribute__((ext_vector_type(4)));
typedef float    v8f  __attribute__((ext_vector_type(8)));

// D = A(16x32 f16) * B(32x16 f16) + C(16x16 f32)
__device__ inline v8f wmma16(v16h a, v16h b, v8f c) {
  return __builtin_amdgcn_wmma_f32_16x16x32_f16(
      /*neg_a=*/false, a, /*neg_b=*/false, b,
      /*c_mod=*/(short)0, c, /*reuse_a=*/false, /*reuse_b=*/false);
}

// A-matrix 16x32 (MxK) f16 fragment (ISA 7.12.2):
//   lane: m = l&15 ; VGPR quad h covers K = 16h + 8*(l>=16) + {0..7}
__device__ inline v16h load_frag_a16(const _Float16* base, int stride) {
  const int l = threadIdx.x & 31;
  const int row = l & 15, hl = l >> 4;
  const _Float16* p = base + (size_t)row * stride;
  v16h f;
#pragma unroll
  for (int h = 0; h < 2; ++h) {
    const int kb = 16 * h + 8 * hl;
    v8h c = *(const v8h*)(p + kb);   // 16B contiguous
#pragma unroll
    for (int j = 0; j < 8; ++j) f[8 * h + j] = c[j];
  }
  return f;
}

// B-matrix 32x16 (KxN) f16 fragment from n-major memory (base[n*stride+k]):
//   lane n = l&15; element s<8 holds K=8*hl+s, s>=8 holds K=16+8*hl+(s-8).
__device__ inline v16h load_frag_b16(const _Float16* base, int stride) {
  const int l = threadIdx.x & 31;
  const int n = l & 15, hl = l >> 4;
  const _Float16* p = base + (size_t)n * stride + 8 * hl;
  v16h f;
  v8h c0 = *(const v8h*)(p);
  v8h c1 = *(const v8h*)(p + 16);
#pragma unroll
  for (int j = 0; j < 8; ++j) { f[j] = c0[j]; f[8 + j] = c1[j]; }
  return f;
}

// ---------------------------------------------------------------------------
// Pack kernels (run once; cost amortized over the whole GEMM grid)
// ---------------------------------------------------------------------------
__global__ __launch_bounds__(256)
void cvt_x_kernel(const float* __restrict__ x, _Float16* __restrict__ Xh) {
  const size_t i = ((size_t)blockIdx.x * 256 + threadIdx.x) * 4;
  const float4 v = *(const float4*)(x + i);
  v4h o = { (_Float16)v.x, (_Float16)v.y, (_Float16)v.z, (_Float16)v.w };
  *(v4h*)(Xh + i) = o;
}

// W f32 [k][n] row-major  ->  Wt f16 [n][k] (n-major), tiled through LDS.
__global__ __launch_bounds__(256)
void transpose_w_kernel(const float* __restrict__ W0, const float* __restrict__ W1,
                        const float* __restrict__ W2, const float* __restrict__ W3,
                        _Float16* __restrict__ T0, _Float16* __restrict__ T1,
                        _Float16* __restrict__ T2, _Float16* __restrict__ T3) {
  __shared__ __align__(16) _Float16 tile[64][65];
  const int z = blockIdx.z;
  const float* W = (z == 0) ? W0 : (z == 1) ? W1 : (z == 2) ? W2 : W3;
  _Float16*    T = (z == 0) ? T0 : (z == 1) ? T1 : (z == 2) ? T2 : T3;
  const int k0 = blockIdx.x * 64, n0 = blockIdx.y * 64;
  const int t = threadIdx.x;
#pragma unroll
  for (int i = 0; i < 16; ++i) {
    const int idx = i * 256 + t;           // 0..4095
    const int kk = idx >> 6, nn = idx & 63;
    tile[kk][nn] = (_Float16)W[(size_t)(k0 + kk) * EMB + n0 + nn];
  }
  __syncthreads();
#pragma unroll
  for (int i = 0; i < 16; ++i) {
    const int idx = i * 256 + t;
    const int nn = idx >> 6, kk = idx & 63;
    T[(size_t)(n0 + nn) * EMB + k0 + kk] = tile[kk][nn];
  }
}

// ---------------------------------------------------------------------------
// Kernel 1: QKV projection, LDS-free WMMA GEMM.
//   grid = (MROWS/64, EMB/128, 3), block = 128 (4 waves, each 32x64)
// ---------------------------------------------------------------------------
__global__ __launch_bounds__(128)
void gemm_qkv_kernel(const _Float16* __restrict__ Xh,
                     const _Float16* __restrict__ Wqt, const _Float16* __restrict__ Wkt,
                     const _Float16* __restrict__ Wvt,
                     const float* __restrict__ bq, const float* __restrict__ bk,
                     const float* __restrict__ bv,
                     _Float16* __restrict__ Qh, _Float16* __restrict__ Kh,
                     _Float16* __restrict__ Vt) {
  const int z = blockIdx.z;
  const _Float16* Wt   = (z == 0) ? Wqt : ((z == 1) ? Wkt : Wvt);
  const float*    bias = (z == 0) ? bq  : ((z == 1) ? bk  : bv);

  const int m0 = blockIdx.x * 64;
  const int n0 = blockIdx.y * 128;
  const int t  = threadIdx.x;
  const int w  = t >> 5, l = t & 31, ln = l & 15, hl = l >> 4;
  const int wm = (w & 1) * 32, wn = (w >> 1) * 64;

  v8f acc[2][4] = {};

  for (int kt = 0; kt < EMB / 32; ++kt) {
    const int k0 = kt * 32;
    const v16h a0 = load_frag_a16(Xh + (size_t)(m0 + wm)      * EMB + k0, EMB);
    const v16h a1 = load_frag_a16(Xh + (size_t)(m0 + wm + 16) * EMB + k0, EMB);
#pragma unroll
    for (int ni = 0; ni < 4; ++ni) {
      const v16h b = load_frag_b16(Wt + (size_t)(n0 + wn + ni * 16) * EMB + k0, EMB);
      acc[0][ni] = wmma16(a0, b, acc[0][ni]);
      acc[1][ni] = wmma16(a1, b, acc[1][ni]);
    }
  }

  // epilogue: bias + scatter into head-major f16 layouts
#pragma unroll
  for (int mi = 0; mi < 2; ++mi)
#pragma unroll
    for (int ni = 0; ni < 4; ++ni)
#pragma unroll
      for (int r = 0; r < 8; ++r) {
        const int m = m0 + wm + mi * 16 + r + 8 * hl;   // row in [0, B*S)
        const int n = n0 + wn + ni * 16 + ln;           // col in [0, E)
        const float val = acc[mi][ni][r] + bias[n];
        const int bb = m >> 11, s = m & (SEQ - 1);
        const int hh = n >> 6, d = n & (HDIM - 1);
        if (z < 2) {
          _Float16* dst = (z == 0) ? Qh : Kh;
          dst[(((size_t)bb * HEADS + hh) * SEQ + s) * HDIM + d] = (_Float16)val;
        } else {
          Vt[(((size_t)bb * HEADS + hh) * HDIM + d) * SEQ + s] = (_Float16)val;
        }
      }
}

// ---------------------------------------------------------------------------
// Kernel 2: flash attention.  grid = (SEQ/64, HEADS, BATCH), block = 128
// Each wave owns 16 query rows; streams key blocks of 32.
// ---------------------------------------------------------------------------
__global__ __launch_bounds__(128)
void attn_kernel(const _Float16* __restrict__ Qh, const _Float16* __restrict__ Kh,
                 const _Float16* __restrict__ Vt, _Float16* __restrict__ Oh) {
  __shared__ __align__(16) _Float16 pbuf[4][16][32];  // per-wave P tile (m x key)

  const int t = threadIdx.x, w = t >> 5, l = t & 31, ln = l & 15, hl = l >> 4;
  const int bb = blockIdx.z, hh = blockIdx.y;
  const size_t bh = (size_t)bb * HEADS + hh;
  const _Float16* Q = Qh + bh * SEQ * HDIM;
  const _Float16* K = Kh + bh * SEQ * HDIM;
  const _Float16* V = Vt + bh * (size_t)HDIM * SEQ;   // [d][s]
  const int q0 = blockIdx.x * 64 + w * 16;

  // Q fragments for this wave's 16 rows (D=64 -> two K=32 steps), kept resident
  const v16h aq0 = load_frag_a16(Q + (size_t)q0 * HDIM, HDIM);
  const v16h aq1 = load_frag_a16(Q + (size_t)q0 * HDIM + 32, HDIM);

  v8f acc[4] = {};           // O accumulator: 16 rows x 64 cols
  float rmax[8], rsum[8];    // per-lane rows m = r + 8*hl (matches C layout)
#pragma unroll
  for (int r = 0; r < 8; ++r) { rmax[r] = -1e30f; rsum[r] = 0.0f; }

  for (int kb = 0; kb < SEQ / 32; ++kb) {
    const int kbase = kb * 32;

    // scores: S = (Q . K^T) * 1/sqrt(D);  K rows are n-major B operands directly
    v8f sc[2];
#pragma unroll
    for (int nt = 0; nt < 2; ++nt) {
      v16h kb0 = load_frag_b16(K + (size_t)(kbase + nt * 16) * HDIM, HDIM);
      v16h kb1 = load_frag_b16(K + (size_t)(kbase + nt * 16) * HDIM + 32, HDIM);
      v8f c = {};
      c = wmma16(aq0, kb0, c);
      c = wmma16(aq1, kb1, c);
#pragma unroll
      for (int r = 0; r < 8; ++r) c[r] *= 0.125f;   // 1/sqrt(64)
      sc[nt] = c;
    }

    // online softmax: row reductions across the 16 lanes sharing each row
#pragma unroll
    for (int r = 0; r < 8; ++r) {
      float mx = fmaxf(sc[0][r], sc[1][r]);
#pragma unroll
      for (int d = 1; d < 16; d <<= 1) mx = fmaxf(mx, __shfl_xor(mx, d, 32));
      const float nm = fmaxf(rmax[r], mx);
      const float corr = __expf(rmax[r] - nm);
      sc[0][r] = __expf(sc[0][r] - nm);
      sc[1][r] = __expf(sc[1][r] - nm);
      float sum = sc[0][r] + sc[1][r];
#pragma unroll
      for (int d = 1; d < 16; d <<= 1) sum += __shfl_xor(sum, d, 32);
      rsum[r] = rsum[r] * corr + sum;
      rmax[r] = nm;
#pragma unroll
      for (int dt = 0; dt < 4; ++dt) acc[dt][r] *= corr;
    }

    // C-layout -> A-layout via per-wave LDS tile (DS is in-order per wave)
#pragma unroll
    for (int nt = 0; nt < 2; ++nt)
#pragma unroll
      for (int r = 0; r < 8; ++r)
        pbuf[w][r + 8 * hl][nt * 16 + ln] = (_Float16)sc[nt][r];
    asm volatile("s_wait_dscnt 0" ::: "memory");

    const v16h ap = load_frag_a16(&pbuf[w][0][0], 32);
#pragma unroll
    for (int dt = 0; dt < 4; ++dt) {
      v16h bv = load_frag_b16(V + (size_t)(dt * 16) * SEQ + kbase, SEQ);
      acc[dt] = wmma16(ap, bv, acc[dt]);
    }
  }

  // normalize and write O as f16 [B*S][E]
#pragma unroll
  for (int dt = 0; dt < 4; ++dt)
#pragma unroll
    for (int r = 0; r < 8; ++r) {
      const int m = q0 + r + 8 * hl;
      const float o = acc[dt][r] / rsum[r];
      Oh[((size_t)bb * SEQ + m) * EMB + hh * HDIM + dt * 16 + ln] = (_Float16)o;
    }
}

// ---------------------------------------------------------------------------
// Kernel 3: output projection, LDS-free WMMA GEMM.
//   grid = (MROWS/64, EMB/128), block = 128 (4 waves, each 32x64)
// ---------------------------------------------------------------------------
__global__ __launch_bounds__(128)
void gemm_out_kernel(const _Float16* __restrict__ Oh, const _Float16* __restrict__ Wot,
                     const float* __restrict__ bo, float* __restrict__ out) {
  const int m0 = blockIdx.x * 64;
  const int n0 = blockIdx.y * 128;
  const int t  = threadIdx.x;
  const int w  = t >> 5, l = t & 31, ln = l & 15, hl = l >> 4;
  const int wm = (w & 1) * 32, wn = (w >> 1) * 64;

  v8f acc[2][4] = {};

  for (int kt = 0; kt < EMB / 32; ++kt) {
    const int k0 = kt * 32;
    const v16h a0 = load_frag_a16(Oh + (size_t)(m0 + wm)      * EMB + k0, EMB);
    const v16h a1 = load_frag_a16(Oh + (size_t)(m0 + wm + 16) * EMB + k0, EMB);
#pragma unroll
    for (int ni = 0; ni < 4; ++ni) {
      const v16h b = load_frag_b16(Wot + (size_t)(n0 + wn + ni * 16) * EMB + k0, EMB);
      acc[0][ni] = wmma16(a0, b, acc[0][ni]);
      acc[1][ni] = wmma16(a1, b, acc[1][ni]);
    }
  }

#pragma unroll
  for (int mi = 0; mi < 2; ++mi)
#pragma unroll
    for (int ni = 0; ni < 4; ++ni)
#pragma unroll
      for (int r = 0; r < 8; ++r) {
        const int m = m0 + wm + mi * 16 + r + 8 * hl;
        const int n = n0 + wn + ni * 16 + ln;
        out[(size_t)m * EMB + n] = acc[mi][ni][r] + bo[n];
      }
}

// ---------------------------------------------------------------------------
extern "C" void kernel_launch(void* const* d_in, const int* in_sizes, int n_in,
                              void* d_out, int out_size, void* d_ws, size_t ws_size,
                              hipStream_t stream) {
  const float* x  = (const float*)d_in[0];
  const float* Wq = (const float*)d_in[1];
  const float* bq = (const float*)d_in[2];
  const float* Wk = (const float*)d_in[3];
  const float* bk = (const float*)d_in[4];
  const float* Wv = (const float*)d_in[5];
  const float* bv = (const float*)d_in[6];
  const float* Wo = (const float*)d_in[7];
  const float* bo = (const float*)d_in[8];
  float* out = (float*)d_out;

  _Float16* ws = (_Float16*)d_ws;
  const size_t XSZ = (size_t)MROWS * EMB;     // 4,194,304 halfs (8 MB)
  const size_t WSZ = (size_t)EMB * EMB;       // 1,048,576 halfs (2 MB)
  _Float16* Xh  = ws;
  _Float16* Wqt = Xh  + XSZ;
  _Float16* Wkt = Wqt + WSZ;
  _Float16* Wvt = Wkt + WSZ;
  _Float16* Wot = Wvt + WSZ;
  _Float16* Qh  = Wot + WSZ;                  // [B,H,S,D]
  _Float16* Kh  = Qh  + XSZ;                  // [B,H,S,D]
  _Float16* Vth = Kh  + XSZ;                  // [B,H,D,S]
  _Float16* Oh  = Vth + XSZ;                  // [B*S][E]   (total 48 MB)

  // 0) pack operands to f16 (x flat; weights transposed to n-major)
  cvt_x_kernel<<<dim3(XSZ / (256 * 4)), dim3(256), 0, stream>>>(x, Xh);
  transpose_w_kernel<<<dim3(EMB / 64, EMB / 64, 4), dim3(256), 0, stream>>>(
      Wq, Wk, Wv, Wo, Wqt, Wkt, Wvt, Wot);

  // 1) QKV projections
  gemm_qkv_kernel<<<dim3(MROWS / 64, EMB / 128, 3), dim3(128), 0, stream>>>(
      Xh, Wqt, Wkt, Wvt, bq, bk, bv, Qh, Kh, Vth);

  // 2) attention
  attn_kernel<<<dim3(SEQ / 64, HEADS, BATCH), dim3(128), 0, stream>>>(Qh, Kh, Vth, Oh);

  // 3) output projection
  gemm_out_kernel<<<dim3(MROWS / 64, EMB / 128), dim3(128), 0, stream>>>(Oh, Wot, bo, out);
}
